// PatchNonlocalPool_77919296684787
// MI455X (gfx1250) — matrix operand order repaired and checked by reading, not compiled, not run on hardware
//
#include <hip/hip_runtime.h>

typedef __attribute__((ext_vector_type(16))) _Float16 v16h;
typedef __attribute__((ext_vector_type(8)))  _Float16 v8h;
typedef __attribute__((ext_vector_type(8)))  float    v8f;

#define HP 264   // ushort pitch per patch row in the half planes (528 B = 132 words == 4 mod 64)
#define GS 65    // float pitch per Gram row (conflict-free column sweeps)

#define AH_BYTES (64 * HP * 2)                 // 33792 B per half plane
#define G_OFF    (2 * AH_BYTES)                // Gram offset
#define Y1_OFF   (G_OFF + 64 * GS * 4)         // y1 offset
#define SMEM_BYTES (Y1_OFF + 64 * 4)           // 84480 B

// v = (float)hi + (float)lo with lo = f16(v - f32(hi)); residual error ~2^-22 relative.
__device__ __forceinline__ void split_hilo(float v, unsigned short& h, unsigned short& l) {
  _Float16 hh = (_Float16)v;
  float hf = (float)hh;
  _Float16 ll = (_Float16)(v - hf);
  h = __builtin_bit_cast(unsigned short, hh);
  l = __builtin_bit_cast(unsigned short, ll);
}

__device__ __forceinline__ float dword_pair_sum(unsigned u) {
  return (float)__builtin_bit_cast(_Float16, (unsigned short)(u & 0xffffu))
       + (float)__builtin_bit_cast(_Float16, (unsigned short)(u >> 16));
}

// One block per (b, n) slice: patchify -> Gram via WMMA (hi/lo split) -> softmax -> y2.
__global__ __launch_bounds__(256) void patch_affinity_kernel(
    const float* __restrict__ x, float* __restrict__ y2buf) {
  extern __shared__ char smem[];
  unsigned short* Ah = (unsigned short*)smem;               // [64][HP] f16 hi plane
  unsigned short* Al = (unsigned short*)(smem + AH_BYTES);  // [64][HP] f16 lo plane
  float*          G  = (float*)(smem + G_OFF);              // [64][GS] Gram (and scratch)
  float*          y1 = (float*)(smem + Y1_OFF);             // [64] patch means

  const int slice = blockIdx.x;
  const int t = threadIdx.x;
  const float4* src = (const float4*)(x + (size_t)slice * 16384);

  // Phase 0: stream the 64 KB slice (coalesced b128 loads), split into hi/lo half planes,
  // stored patch-major so WMMA fragments are contiguous b128 runs.
  for (int i = 0; i < 16; ++i) {
    int idx4 = t + 256 * i;
    float4 v = src[idx4];
    int lin = idx4 * 4;
    int rg = lin >> 7, cg = lin & 127;           // row/col in 128x128
    int p = (rg >> 4) * 8 + (cg >> 4);           // patch id 0..63
    int e = (rg & 15) * 16 + (cg & 15);          // element 0..255 (multiple of 4 here)
    unsigned short h0, h1, h2, h3, l0, l1, l2, l3;
    split_hilo(v.x, h0, l0); split_hilo(v.y, h1, l1);
    split_hilo(v.z, h2, l2); split_hilo(v.w, h3, l3);
    uint2 hv, lv;
    hv.x = (unsigned)h0 | ((unsigned)h1 << 16); hv.y = (unsigned)h2 | ((unsigned)h3 << 16);
    lv.x = (unsigned)l0 | ((unsigned)l1 << 16); lv.y = (unsigned)l2 | ((unsigned)l3 << 16);
    *(uint2*)(Ah + p * HP + e) = hv;             // 8-byte aligned ds_store_b64
    *(uint2*)(Al + p * HP + e) = lv;
  }
  __syncthreads();

  // Phase 1: per-patch means y1 (G used as reduction scratch before WMMA overwrites it).
  {
    int p = t >> 2, q = t & 3;
    float s = 0.f;
    for (int e = q * 64; e < q * 64 + 64; e += 4) {
      uint2 hv = *(const uint2*)(Ah + p * HP + e);
      uint2 lv = *(const uint2*)(Al + p * HP + e);
      s += dword_pair_sum(hv.x) + dword_pair_sum(hv.y)
         + dword_pair_sum(lv.x) + dword_pair_sum(lv.y);
    }
    G[t] = s;
  }
  __syncthreads();
  if (t < 64) y1[t] = (G[4*t] + G[4*t+1] + G[4*t+2] + G[4*t+3]) * (1.f / 256.f);
  __syncthreads();

  // Phase 2: Gram = A * A^T (64x64, K=256), 16 tiles of 16x16, 2 per wave.
  // D accumulates hi*hi + hi*lo + lo*hi in fp32 -> ~fp32-accurate logits.
  const int wv = t >> 5, lane = t & 31;
  const int g = lane >> 4, lm = lane & 15;
  for (int tile = wv * 2; tile < wv * 2 + 2; ++tile) {
    const int ti = tile >> 2, tj = tile & 3;
    const int rowA = (ti * 16 + lm) * HP;   // A fragment rows: M = ti*16 + lm
    const int rowB = (tj * 16 + lm) * HP;   // B fragment "rows": N = tj*16 + lm (A^T gather)
    v8f c = {};
    for (int k0 = 0; k0 < 256; k0 += 32) {
      // A(16x32): VGPRs 0-3 = K ka..ka+7, VGPRs 4-7 = K ka+16..ka+23 (two b128 loads)
      const int ka = k0 + 8 * g;
      v8h ah0 = *(const v8h*)(Ah + rowA + ka);
      v8h ah1 = *(const v8h*)(Ah + rowA + ka + 16);
      v8h al0 = *(const v8h*)(Al + rowA + ka);
      v8h al1 = *(const v8h*)(Al + rowA + ka + 16);
      // B(32x16): VGPRs 0-3 = K kb..kb+7, VGPRs 4-7 = K kb+16..kb+23 (two b128 loads)
      const int kb = k0 + 8 * g;
      v8h bh0 = *(const v8h*)(Ah + rowB + kb);
      v8h bh1 = *(const v8h*)(Ah + rowB + kb + 16);
      v8h bl0 = *(const v8h*)(Al + rowB + kb);
      v8h bl1 = *(const v8h*)(Al + rowB + kb + 16);
      v16h A_hi = __builtin_shufflevector(ah0, ah1, 0,1,2,3,4,5,6,7,8,9,10,11,12,13,14,15);
      v16h A_lo = __builtin_shufflevector(al0, al1, 0,1,2,3,4,5,6,7,8,9,10,11,12,13,14,15);
      v16h B_hi = __builtin_shufflevector(bh0, bh1, 0,1,2,3,4,5,6,7,8,9,10,11,12,13,14,15);
      v16h B_lo = __builtin_shufflevector(bl0, bl1, 0,1,2,3,4,5,6,7,8,9,10,11,12,13,14,15);
      c = __builtin_amdgcn_wmma_f32_16x16x32_f16(false, A_hi, false, B_hi, (short)0, c, false, false);
      c = __builtin_amdgcn_wmma_f32_16x16x32_f16(false, A_hi, false, B_lo, (short)0, c, false, false);
      c = __builtin_amdgcn_wmma_f32_16x16x32_f16(false, A_lo, false, B_hi, (short)0, c, false, false);
    }
    // C/D layout: VGPR r holds row M = r + 8*g, col N = lm
    #pragma unroll
    for (int r = 0; r < 8; ++r)
      G[(ti * 16 + r + 8 * g) * GS + tj * 16 + lm] = c[r];
  }
  __syncthreads();

  // Phase 3: row softmax fused with dot(y1) -> y2.
  if (t < 64) {
    const float* row = G + t * GS;
    float m = -3.4e38f;
    for (int j = 0; j < 64; ++j) m = fmaxf(m, row[j]);
    float s = 0.f, acc = 0.f;
    for (int j = 0; j < 64; ++j) {
      float e = __expf(row[j] - m);
      s += e;
      acc += e * y1[j];
    }
    y2buf[slice * 64 + t] = acc / s;
  }
}

// mean(conv3x3_SAME(y2)) == (1/64) * sum_{ic,kh,kw} w[oc,ic,kh,kw] * T[b,ic,kh,kw],
// where T is the clipped-window sum of the 8x8 map. One block per batch b.
__global__ __launch_bounds__(128) void conv_pool_kernel(
    const float* __restrict__ conv_w, const float* __restrict__ y2buf,
    float* __restrict__ out) {
  __shared__ float Tl[128 * 9];
  const int b = blockIdx.x;
  const int t = threadIdx.x;

  {
    const int ic = t;
    const float* y = y2buf + ((size_t)b * 128 + ic) * 64;
    float acc[9] = {0.f, 0.f, 0.f, 0.f, 0.f, 0.f, 0.f, 0.f, 0.f};
    for (int r = 0; r < 8; ++r) {
      float full = 0.f;
      #pragma unroll
      for (int cc = 0; cc < 8; ++cc) full += y[r * 8 + cc];
      float left  = full - y[r * 8 + 7];   // cols 0..6  (kw = 0)
      float right = full - y[r * 8 + 0];   // cols 1..7  (kw = 2)
      if (r <= 6) { acc[0] += left; acc[1] += full; acc[2] += right; }  // kh = 0
                  { acc[3] += left; acc[4] += full; acc[5] += right; }  // kh = 1
      if (r >= 1) { acc[6] += left; acc[7] += full; acc[8] += right; }  // kh = 2
    }
    #pragma unroll
    for (int k = 0; k < 9; ++k) Tl[ic * 9 + k] = acc[k];
  }
  __syncthreads();

  {
    const int oc = t;
    const float* wr = conv_w + (size_t)oc * 1152;   // [ic][kh][kw] flat
    float s = 0.f;
    for (int i = 0; i < 1152; ++i) s += wr[i] * Tl[i];
    out[b * 128 + oc] = s * (1.f / 64.f);
  }
}

extern "C" void kernel_launch(void* const* d_in, const int* in_sizes, int n_in,
                              void* d_out, int out_size, void* d_ws, size_t ws_size,
                              hipStream_t stream) {
  const float* x = (const float*)d_in[0];   // [32,128,128,128] f32
  const float* w = (const float*)d_in[1];   // [128,128,3,3] f32
  float* out = (float*)d_out;               // [32,128,1,1] f32
  float* y2buf = (float*)d_ws;              // [4096,64] f32 = 1 MB scratch

  patch_affinity_kernel<<<4096, 256, SMEM_BYTES, stream>>>(x, y2buf);
  conv_pool_kernel<<<32, 128, 0, stream>>>(w, y2buf, out);
}
